// MultiHeadAttention_70171175682641
// MI455X (gfx1250) — matrix operand order repaired and checked by reading, not compiled, not run on hardware
//
#include <hip/hip_runtime.h>

// ---------------------------------------------------------------------------
// Types for CDNA5 WMMA (wave32): v_wmma_f32_16x16x32_bf16
// ---------------------------------------------------------------------------
typedef __attribute__((ext_vector_type(16))) __bf16 v16bf;
typedef __attribute__((ext_vector_type(8)))  __bf16 v8bf;
typedef __attribute__((ext_vector_type(4)))  __bf16 v4bf;
typedef __attribute__((ext_vector_type(8)))  float  v8f;
typedef __attribute__((ext_vector_type(4)))  unsigned int u32x4;
typedef __attribute__((ext_vector_type(8)))  int          i32x8;
typedef __attribute__((ext_vector_type(4)))  int          i32x4;

// Problem constants
#define BB 4
#define TT 2048
#define DD 1024
#define HH 16
#define HD 64
#define M_TOK (BB * TT)          // 8192 tokens

__device__ __forceinline__ v8f wmma_bf16(v16bf a, v16bf b, v8f c) {
  // (neg_a, A, neg_b, B, c_mod, C, reuse_a, reuse_b)
  return __builtin_amdgcn_wmma_f32_16x16x32_bf16(false, a, false, b, (short)0, c,
                                                 false, false);
}

// A fragment (16x32 bf16): lane = row (lane%16); per-lane K values are
//   {kbase..kbase+7} and {kbase+16..kbase+23} with kbase = k0 + 8*(lane>=16).
__device__ __forceinline__ v16bf load_a16(const __bf16* base, int ld, int row, int kbase) {
  const __bf16* p = base + (size_t)row * ld + kbase;
  v8bf lo = *reinterpret_cast<const v8bf*>(p);
  v8bf hi = *reinterpret_cast<const v8bf*>(p + 16);
  v16bf r;
#pragma unroll
  for (int i = 0; i < 8; ++i) { r[i] = lo[i]; r[i + 8] = hi[i]; }
  return r;
}

// B fragment (32x16 bf16): lane = column (lane%16); per-lane K values are
//   {kbase..kbase+7} and {kbase+16..kbase+23} with kbase = k0 + 8*(lane>=16).
// Source must be stored K-contiguous per column, i.e. B^T row-major [N][K].
__device__ __forceinline__ v16bf load_b16(const __bf16* base, int ld, int row, int kbase) {
  const __bf16* p = base + (size_t)row * ld + kbase;
  v8bf lo = *reinterpret_cast<const v8bf*>(p);
  v8bf hi = *reinterpret_cast<const v8bf*>(p + 16);
  v16bf r;
#pragma unroll
  for (int i = 0; i < 8; ++i) { r[i] = lo[i]; r[i + 8] = hi[i]; }
  return r;
}

// ---------------------------------------------------------------------------
// Tensor Data Mover: async DMA of a 2D tile [rows x 32 bf16], row pitch
// ld_elems, from global (tile start = ga) into LDS offset lds_off.
// D# layout per CDNA5 ISA.
// ---------------------------------------------------------------------------
__device__ __forceinline__ void tdm_load_tile_2d(unsigned int lds_off,
                                                 unsigned long long ga,
                                                 int rows, int ld_elems) {
  u32x4 g0;
  g0[0] = 1u;                                            // count=1, user descriptor
  g0[1] = lds_off;                                       // lds_addr [63:32]
  g0[2] = (unsigned int)(ga & 0xFFFFFFFFull);            // global_addr lo
  g0[3] = (unsigned int)((ga >> 32) & 0x01FFFFFFull)     // global_addr hi [120:96]
        | 0x80000000u;                                   // type=2 ("image") [127:126]
  i32x8 g1;
  g1[0] = 0x00010000;            // workgroup_mask=0, data_size=1 (2 bytes)
  g1[1] = 0;                     // atomic_barrier=0, tensor_dim0 lo16 = 0
  g1[2] = 0x00004000;            // tensor_dim0 = 0x40000000 (hi16), dim1 lo16=0
  g1[3] = (32 << 16) | 0x4000;   // tensor_dim1 hi16 = 0x4000, tile_dim0 = 32
  g1[4] = rows & 0xFFFF;         // tile_dim1 = rows, tile_dim2 = 0
  g1[5] = ld_elems;              // tensor_dim0_stride lo32 (elements)
  g1[6] = 0;                     // stride hi16 = 0, dim1_stride lo16 = 0
  g1[7] = 0;
  i32x4 z4; z4[0] = 0; z4[1] = 0; z4[2] = 0; z4[3] = 0;
#if defined(__clang_major__) && __clang_major__ >= 23
  i32x8 z8;
#pragma unroll
  for (int i = 0; i < 8; ++i) z8[i] = 0;
  __builtin_amdgcn_tensor_load_to_lds(g0, g1, z4, z4, z8, 0);
#else
  __builtin_amdgcn_tensor_load_to_lds(g0, g1, z4, z4, 0);
#endif
}

// ---------------------------------------------------------------------------
// Block-cooperative GEMM: block tile 128x256 (8 waves, 2x4 of 64x64 wave
// tiles). A[M,K] and Bt[N,K] staged per 32-wide K-step into LDS by the TDM
// (double-buffered), fragments read from LDS. acc[mi][ni] holds C rows
// (mi*16 + r + 8*half), col (ni*16 + lane%16) of the wave's 64x64 tile.
// ---------------------------------------------------------------------------
__device__ __forceinline__ void block_gemm_tdm(const __bf16* __restrict__ A,
                                               const __bf16* __restrict__ Bt,
                                               int Kdim, int mblk, int nblk,
                                               v8f acc[4][4]) {
  __shared__ __bf16 lA[2][128 * 32];   // 16 KB
  __shared__ __bf16 lB[2][256 * 32];   // 32 KB

  const int lane = threadIdx.x & 31;
  const int wid  = threadIdx.x >> 5;
  const int half = lane >> 4;
  const int l16  = lane & 15;
  const int mw   = (wid >> 2) * 64;    // wave row offset inside block tile
  const int nw   = (wid & 3) * 64;     // wave col offset inside block tile

#pragma unroll
  for (int mi = 0; mi < 4; ++mi)
#pragma unroll
    for (int ni = 0; ni < 4; ++ni)
#pragma unroll
      for (int i = 0; i < 8; ++i) acc[mi][ni][i] = 0.0f;

  const unsigned long long baseA =
      (unsigned long long)(size_t)(A + (size_t)mblk * Kdim);
  const unsigned long long baseB =
      (unsigned long long)(size_t)(Bt + (size_t)nblk * Kdim);
  const unsigned int offA[2] = {(unsigned int)(size_t)&lA[0][0],
                                (unsigned int)(size_t)&lA[1][0]};
  const unsigned int offB[2] = {(unsigned int)(size_t)&lB[0][0],
                                (unsigned int)(size_t)&lB[1][0]};

  const int nsteps = Kdim >> 5;
  if (wid == 0) {                       // one wave drives the TDM
    tdm_load_tile_2d(offA[0], baseA, 128, Kdim);
    tdm_load_tile_2d(offB[0], baseB, 256, Kdim);
  }

  for (int i = 0; i < nsteps; ++i) {
    const int buf = i & 1;
    if (wid == 0) __builtin_amdgcn_s_wait_tensorcnt(0);  // buf ready (issuer)
    __syncthreads();                                     // publish to all waves
    if (wid == 0 && (i + 1) < nsteps) {                  // prefetch next K-step
      const unsigned long long koff = (unsigned long long)(i + 1) * 64; // 32*2B
      tdm_load_tile_2d(offA[buf ^ 1], baseA + koff, 128, Kdim);
      tdm_load_tile_2d(offB[buf ^ 1], baseB + koff, 256, Kdim);
    }

    const __bf16* tA = &lA[buf][0];
    const __bf16* tB = &lB[buf][0];
    v16bf a[4], b[4];
#pragma unroll
    for (int mi = 0; mi < 4; ++mi)
      a[mi] = load_a16(tA, 32, mw + mi * 16 + l16, half * 8);
#pragma unroll
    for (int ni = 0; ni < 4; ++ni)
      b[ni] = load_b16(tB, 32, nw + ni * 16 + l16, half * 8);
#pragma unroll
    for (int mi = 0; mi < 4; ++mi)
#pragma unroll
      for (int ni = 0; ni < 4; ++ni)
        acc[mi][ni] = wmma_bf16(a[mi], b[ni], acc[mi][ni]);

    __syncthreads();   // all reads of buf done before it is DMA-overwritten
  }
}

// ---------------------------------------------------------------------------
// Elementwise fp32 -> bf16 (vector x4)
// ---------------------------------------------------------------------------
__global__ void k_f32_to_bf16(const float* __restrict__ in, __bf16* __restrict__ out,
                              int n4) {
  int i = blockIdx.x * blockDim.x + threadIdx.x;
  if (i >= n4) return;
  float4 v = reinterpret_cast<const float4*>(in)[i];
  v4bf o;
  o[0] = (__bf16)v.x; o[1] = (__bf16)v.y; o[2] = (__bf16)v.z; o[3] = (__bf16)v.w;
  reinterpret_cast<v4bf*>(out)[i] = o;
}

// Transpose fp32 [K][N] -> bf16 [N][K]
__global__ void k_w_transpose_bf16(const float* __restrict__ in,
                                   __bf16* __restrict__ out, int K, int N) {
  int idx = blockIdx.x * blockDim.x + threadIdx.x;
  if (idx >= K * N) return;
  int n = idx / K;
  int k = idx - n * K;
  out[idx] = (__bf16)in[(size_t)k * N + n];
}

// ---------------------------------------------------------------------------
// QKV projection: X[8192,1024]bf16 * WqkvT[3072,1024]bf16 ->
//   Q,K scattered to [B,H,T,64] bf16 ; V scattered transposed to [B,H,64,T].
// Block = 256 threads = 8 waves (2x4), block tile 128x256. Grid (64, 12).
// ---------------------------------------------------------------------------
__global__ __launch_bounds__(256) void k_gemm_qkv(const __bf16* __restrict__ X,
                                                  const __bf16* __restrict__ Wt,
                                                  __bf16* __restrict__ qbuf,
                                                  __bf16* __restrict__ kbuf,
                                                  __bf16* __restrict__ vtbuf) {
  const int mblk = blockIdx.x * 128;
  const int nblk = blockIdx.y * 256;
  v8f acc[4][4];
  block_gemm_tdm(X, Wt, DD, mblk, nblk, acc);

  const int wid  = threadIdx.x >> 5;
  const int lane = threadIdx.x & 31;
  const int half = lane >> 4;
  const int l16  = lane & 15;
  const int m0 = mblk + (wid >> 2) * 64;
  const int n0 = nblk + (wid & 3) * 64;
#pragma unroll
  for (int mi = 0; mi < 4; ++mi) {
#pragma unroll
    for (int ni = 0; ni < 4; ++ni) {
      const int e     = n0 + ni * 16 + l16;
      const int which = e >> 10;       // 0=q, 1=k, 2=v
      const int din   = e & 1023;
      const int hh    = din >> 6;
      const int dd    = din & 63;
#pragma unroll
      for (int r = 0; r < 8; ++r) {
        const int m = m0 + mi * 16 + half * 8 + r;
        const int b = m >> 11;
        const int t = m & 2047;
        const size_t bh = (size_t)(b * HH + hh);
        const __bf16 v = (__bf16)acc[mi][ni][r];
        if (which == 0)      qbuf[(bh * TT + t) * HD + dd] = v;
        else if (which == 1) kbuf[(bh * TT + t) * HD + dd] = v;
        else                 vtbuf[(bh * HD + dd) * TT + t] = v;
      }
    }
  }
}

// ---------------------------------------------------------------------------
// Flash attention (causal, online softmax). One wave = one 16-row Q block.
// Streams 32-key blocks: S = Q*K^T (4 wmma), mask+softmax, P relayout via
// per-wave LDS slice (DS in-order within a wave -> no barrier; waves have
// divergent trip counts so workgroup barriers are forbidden here),
// O += P*V (4 wmma). Output written bf16 as [B,T,D] (d = h*64 + hd).
// ---------------------------------------------------------------------------
__global__ __launch_bounds__(256) void k_attn(const __bf16* __restrict__ Q,
                                              const __bf16* __restrict__ Km,
                                              const __bf16* __restrict__ Vt,
                                              __bf16* __restrict__ Obuf) {
  __shared__ __bf16 plds[8][16 * 32];   // per-wave P tile (16 q-rows x 32 keys)

  const int wid  = threadIdx.x >> 5;
  const int lane = threadIdx.x & 31;
  const int half = lane >> 4;
  const int l16  = lane & 15;

  const int gq = blockIdx.x * 8 + wid;    // global q-block id
  const int bh = gq >> 7;                 // T/16 = 128 q-blocks per (b,h)
  const int q0 = (gq & 127) << 4;

  const __bf16* Qp = Q  + (size_t)bh * TT * HD;
  const __bf16* Kp = Km + (size_t)bh * TT * HD;
  const __bf16* Vp = Vt + (size_t)bh * HD * TT;

  // Q A-fragments for the two K-steps over head_dim (64 = 2 x 32)
  v16bf qf[2];
#pragma unroll
  for (int s = 0; s < 2; ++s)
    qf[s] = load_a16(Qp, HD, q0 + l16, s * 32 + half * 8);

  v8f o[4];
#pragma unroll
  for (int di = 0; di < 4; ++di)
#pragma unroll
    for (int i = 0; i < 8; ++i) o[di][i] = 0.0f;

  float rm[8], rl[8];
#pragma unroll
  for (int r = 0; r < 8; ++r) { rm[r] = -__builtin_inff(); rl[r] = 0.0f; }

  const float scale = 1.0f / 32.0f;       // 1/sqrt(D), D=1024
  const int qa_base = q0 + half * 8;

  for (int kb = 0; kb < q0 + 16; kb += 32) {
    // --- S = Q * K^T  (rows = q, cols = 32 keys as two 16-col tiles) ---
    v8f s0, s1;
#pragma unroll
    for (int i = 0; i < 8; ++i) { s0[i] = 0.0f; s1[i] = 0.0f; }
#pragma unroll
    for (int st = 0; st < 2; ++st) {
      v16bf kf0 = load_b16(Kp, HD, kb + l16,      st * 32 + half * 8);
      v16bf kf1 = load_b16(Kp, HD, kb + 16 + l16, st * 32 + half * 8);
      s0 = wmma_bf16(qf[st], kf0, s0);
      s1 = wmma_bf16(qf[st], kf1, s1);
    }

    // --- scale, causal mask, online softmax (rows live in 16-lane halves) ---
#pragma unroll
    for (int r = 0; r < 8; ++r) {
      const int qa = qa_base + r;
      float v0 = s0[r] * scale; if (kb + l16 > qa)      v0 = -__builtin_inff();
      float v1 = s1[r] * scale; if (kb + 16 + l16 > qa) v1 = -__builtin_inff();

      float bm = fmaxf(v0, v1);
      bm = fmaxf(bm, __shfl_xor(bm, 1, 32));
      bm = fmaxf(bm, __shfl_xor(bm, 2, 32));
      bm = fmaxf(bm, __shfl_xor(bm, 4, 32));
      bm = fmaxf(bm, __shfl_xor(bm, 8, 32));

      const float nm = fmaxf(rm[r], bm);
      const float sc = __expf(rm[r] - nm);
      const float p0 = __expf(v0 - nm);
      const float p1 = __expf(v1 - nm);
      float ps = p0 + p1;
      ps += __shfl_xor(ps, 1, 32);
      ps += __shfl_xor(ps, 2, 32);
      ps += __shfl_xor(ps, 4, 32);
      ps += __shfl_xor(ps, 8, 32);
      rl[r] = rl[r] * sc + ps;
      rm[r] = nm;
#pragma unroll
      for (int di = 0; di < 4; ++di) o[di][r] *= sc;

      // stash P (C-layout) into LDS as bf16 at [row][key]
      plds[wid][(half * 8 + r) * 32 + l16]      = (__bf16)p0;
      plds[wid][(half * 8 + r) * 32 + 16 + l16] = (__bf16)p1;
    }
    asm volatile("" ::: "memory");  // keep DS loads below DS stores (HW is in-order per wave)

    // --- O += P * V : reload P in A-layout, V^T rows give B fragments ---
    {
      v16bf pf = load_a16(&plds[wid][0], 32, l16, half * 8);
#pragma unroll
      for (int di = 0; di < 4; ++di) {
        v16bf vf = load_b16(Vp, TT, di * 16 + l16, kb + half * 8);
        o[di] = wmma_bf16(pf, vf, o[di]);
      }
    }
  }

  // --- epilogue: normalize and store bf16 [B,T,D] ---
  const int b = bh >> 4;
  const int h = bh & 15;
#pragma unroll
  for (int di = 0; di < 4; ++di)
#pragma unroll
    for (int r = 0; r < 8; ++r) {
      const int t   = q0 + half * 8 + r;
      const int col = h * HD + di * 16 + l16;
      Obuf[((size_t)(b * TT + t)) * DD + col] = (__bf16)(o[di][r] / rl[r]);
    }
}

// ---------------------------------------------------------------------------
// Output projection: Obuf[8192,1024]bf16 * WprojT[1024,1024]bf16 -> fp32 out
// ---------------------------------------------------------------------------
__global__ __launch_bounds__(256) void k_gemm_proj(const __bf16* __restrict__ A,
                                                   const __bf16* __restrict__ Wt,
                                                   float* __restrict__ out) {
  const int mblk = blockIdx.x * 128;
  const int nblk = blockIdx.y * 256;
  v8f acc[4][4];
  block_gemm_tdm(A, Wt, DD, mblk, nblk, acc);

  const int wid  = threadIdx.x >> 5;
  const int lane = threadIdx.x & 31;
  const int half = lane >> 4;
  const int l16  = lane & 15;
  const int m0 = mblk + (wid >> 2) * 64;
  const int n0 = nblk + (wid & 3) * 64;
#pragma unroll
  for (int mi = 0; mi < 4; ++mi)
#pragma unroll
    for (int ni = 0; ni < 4; ++ni)
#pragma unroll
      for (int r = 0; r < 8; ++r) {
        const int m = m0 + mi * 16 + half * 8 + r;
        const int n = n0 + ni * 16 + l16;
        out[(size_t)m * DD + n] = acc[mi][ni][r];
      }
}

// ---------------------------------------------------------------------------
// Host-side launcher
// ---------------------------------------------------------------------------
extern "C" void kernel_launch(void* const* d_in, const int* in_sizes, int n_in,
                              void* d_out, int out_size, void* d_ws, size_t ws_size,
                              hipStream_t stream) {
  (void)in_sizes; (void)n_in; (void)out_size; (void)ws_size;
  const float* x      = (const float*)d_in[0];   // [B,T,D]
  const float* w_qkv  = (const float*)d_in[1];   // [D,3D]
  const float* w_proj = (const float*)d_in[2];   // [D,D]
  float* out = (float*)d_out;                    // [B,T,D]

  // Workspace carve-up (bytes), all 256B aligned by construction.
  char* ws = (char*)d_ws;
  constexpr size_t N_X    = (size_t)M_TOK * DD;         // 8192*1024
  constexpr size_t N_WQ   = (size_t)3 * DD * DD;        // 3072*1024
  constexpr size_t N_WP   = (size_t)DD * DD;            // 1024*1024
  constexpr size_t N_HEAD = (size_t)BB * HH * TT * HD;  // 64*2048*64

  __bf16* xbf    = (__bf16*)ws;                 ws += N_X * 2;
  __bf16* wqkvT  = (__bf16*)ws;                 ws += N_WQ * 2;
  __bf16* wprojT = (__bf16*)ws;                 ws += N_WP * 2;
  __bf16* qbuf   = (__bf16*)ws;                 ws += N_HEAD * 2;
  __bf16* kbuf   = (__bf16*)ws;                 ws += N_HEAD * 2;
  __bf16* vtbuf  = (__bf16*)ws;                 ws += N_HEAD * 2;
  __bf16* obuf   = (__bf16*)ws;                 ws += N_X * 2;

  // 1) x -> bf16
  {
    int n4 = (int)(N_X / 4);
    k_f32_to_bf16<<<(n4 + 255) / 256, 256, 0, stream>>>(x, xbf, n4);
  }
  // 2) weight transposes -> bf16 [N][K]
  {
    int tot = (int)N_WQ;
    k_w_transpose_bf16<<<(tot + 255) / 256, 256, 0, stream>>>(w_qkv, wqkvT, DD, 3 * DD);
  }
  {
    int tot = (int)N_WP;
    k_w_transpose_bf16<<<(tot + 255) / 256, 256, 0, stream>>>(w_proj, wprojT, DD, DD);
  }
  // 3) QKV projection (scatter into attention layouts)
  k_gemm_qkv<<<dim3(M_TOK / 128, (3 * DD) / 256), 256, 0, stream>>>(
      xbf, wqkvT, qbuf, kbuf, vtbuf);
  // 4) flash attention: 8192 q-blocks, 8 waves per block
  k_attn<<<(BB * HH * (TT / 16)) / 8, 256, 0, stream>>>(qbuf, kbuf, vtbuf, obuf);
  // 5) output projection -> fp32
  k_gemm_proj<<<dim3(M_TOK / 128, DD / 256), 256, 0, stream>>>(obuf, wprojT, out);
}
